// Generator_61993557950813
// MI455X (gfx1250) — compile-verified, matrix operand order checked by reading, not run on hardware
//
#include <hip/hip_runtime.h>
#include <hip/hip_bf16.h>
#include <math.h>

// ---------------------------------------------------------------------------
// Seq2seq GRU (B=64,T=1024,V=390,E=H=32) for MI455X / gfx1250, wave32 + WMMA.
//   K1: Wc = enc_Wih0 @ enc_emb_W            (96x390, f16, folds embedding)
//   K2: gx0 = input @ Wc^T + bih0            (WMMA GEMM, M=65536,N=96,K=390)
//   K3: encoder scan, 1 workgroup, 8 waves, per-step 64x96x32 WMMA matmuls
//   K4: decoder scan (argmax feedback), same persistent-workgroup structure,
//       pred MLP + logits via WMMA, 3-pass softmax/argmax through ws scratch.
// ---------------------------------------------------------------------------

typedef __attribute__((ext_vector_type(16))) _Float16 v16h;
typedef __attribute__((ext_vector_type(8)))  float    v8f;

#define BB 64
#define TT 1024
#define VV 390
#define EE 32
#define HH 32
#define G3 96          // 3*H gate width
#define LDWC 416       // padded K stride for Wc (>= 390+26, multiple of 32)
#define STARTTOK 388
#define NOUT ((size_t)BB * TT * VV)

__device__ __forceinline__ float sigmf(float x) { return 1.0f / (1.0f + __expf(-x)); }

// D = A x B + C, A 16x32 f16, B 32x16 f16, C/D 16x16 f32 (codegen-confirmed)
__device__ __forceinline__ v8f wmma32(v16h a, v16h b, v8f c) {
  return __builtin_amdgcn_wmma_f32_16x16x32_f16(false, a, false, b, (short)0, c,
                                                false, false);
}

// K index for element e of the 16xf16 A/B operand, per ISA 16-bit layout:
// lanes 0-15: e0..7 -> K=0..7, e8..15 -> K=16..23; lanes 16-31: +8.
__device__ __forceinline__ int wmma_kidx(int e, int hi) {
  return ((e >> 3) << 4) + (hi << 3) + (e & 7);
}

// A tile (16x32) from row-major f32 global, with K guard (zero pad).
__device__ __forceinline__ v16h load_a_g(const float* p, int ld, int m0, int k0,
                                         int kmax) {
  const int lane = threadIdx.x & 31;
  const int m = m0 + (lane & 15);
  const int hi = lane >> 4;
  v16h a;
#pragma unroll
  for (int e = 0; e < 16; ++e) {
    const int k = k0 + wmma_kidx(e, hi);
    a[e] = (k < kmax) ? (_Float16)p[(size_t)m * ld + k] : (_Float16)0.0f;
  }
  return a;
}

// A tile (16x32) from LDS f32 matrix with ld==32 (hidden states, zb).
__device__ __forceinline__ v16h load_a_lds(const float* sh, int m0) {
  const int lane = threadIdx.x & 31;
  const int m = m0 + (lane & 15);
  const int hi = lane >> 4;
  v16h a;
#pragma unroll
  for (int e = 0; e < 16; ++e) a[e] = (_Float16)sh[m * 32 + wmma_kidx(e, hi)];
  return a;
}

// A tile gathered through an index vector (decoder embedding rows, E=32).
__device__ __forceinline__ v16h load_a_gather(const float* emb, const int* idx,
                                              int m0) {
  const int lane = threadIdx.x & 31;
  const int m = m0 + (lane & 15);
  const int hi = lane >> 4;
  const float* row = emb + (size_t)idx[m] * EE;
  v16h a;
#pragma unroll
  for (int e = 0; e < 16; ++e) a[e] = (_Float16)row[wmma_kidx(e, hi)];
  return a;
}

// B tile (32x16) = W^T from row-major f32 W[N,K]: B(k,n) = W[n*ldk + k].
__device__ __forceinline__ v16h load_b_wT(const float* W, int ldk, int n0,
                                          int k0) {
  const int lane = threadIdx.x & 31;
  const int n = n0 + (lane & 15);
  const int hi = lane >> 4;
  v16h b;
#pragma unroll
  for (int e = 0; e < 16; ++e)
    b[e] = (_Float16)W[(size_t)n * ldk + k0 + wmma_kidx(e, hi)];
  return b;
}

// Same with N guard (pred_W2, N=390 not multiple of 16).
__device__ __forceinline__ v16h load_b_wT_ng(const float* W, int ldk, int n0,
                                             int k0, int nmax) {
  const int lane = threadIdx.x & 31;
  const int n = n0 + (lane & 15);
  const int hi = lane >> 4;
  v16h b;
#pragma unroll
  for (int e = 0; e < 16; ++e)
    b[e] = (n < nmax) ? (_Float16)W[(size_t)n * ldk + k0 + wmma_kidx(e, hi)]
                      : (_Float16)0.0f;
  return b;
}

// B tile from pre-converted f16 weights (Wc), padded ld so no K guard needed.
__device__ __forceinline__ v16h load_b_h16(const _Float16* W, int ldk, int n0,
                                           int k0) {
  const int lane = threadIdx.x & 31;
  const int n = n0 + (lane & 15);
  const int hi = lane >> 4;
  v16h b;
#pragma unroll
  for (int e = 0; e < 16; ++e) b[e] = W[(size_t)n * ldk + k0 + wmma_kidx(e, hi)];
  return b;
}

// Scatter 16x16 f32 D tile into LDS (row-major, given ld).
__device__ __forceinline__ void store_d_lds(float* sh, int ld, int m0, int n0,
                                            v8f d) {
  const int lane = threadIdx.x & 31;
  const int n = n0 + (lane & 15);
  const int hi = lane >> 4;
#pragma unroll
  for (int r = 0; r < 8; ++r) sh[(m0 + r + 8 * hi) * ld + n] = d[r];
}

// ---------------------------------------------------------------------------
// K1: Wc[n][k] = sum_e enc_Wih0[n][e] * enc_emb_W[e][k]  (f16 out, zero-pad k)
// ---------------------------------------------------------------------------
__global__ void k_wc(const float* __restrict__ wih0,
                     const float* __restrict__ wemb,
                     _Float16* __restrict__ wc) {
  const int g = blockIdx.x * blockDim.x + threadIdx.x;
  if (g >= G3 * LDWC) return;
  const int n = g / LDWC, k = g % LDWC;
  float acc = 0.0f;
  if (k < VV) {
#pragma unroll
    for (int e = 0; e < EE; ++e) acc += wih0[n * EE + e] * wemb[e * VV + k];
  }
  wc[g] = (_Float16)acc;
}

// ---------------------------------------------------------------------------
// K2: gx0[t*64+b][0:96] = input[b*T+t][:] @ Wc^T + bih0   (WMMA GEMM)
// One wave per 16x16 output tile; 4096 x 6 tiles.
// ---------------------------------------------------------------------------
__global__ __launch_bounds__(128) void k_gx0(const float* __restrict__ inp,
                                             const _Float16* __restrict__ wc,
                                             const float* __restrict__ bih0,
                                             float* __restrict__ gx0) {
  const int wid = blockIdx.x * (blockDim.x >> 5) + (threadIdx.x >> 5);
  const int mt = wid / 6, nt = wid % 6;
  const int n0 = nt * 16;
  v8f acc = {};
  for (int k0 = 0; k0 < VV; k0 += 32) {
    v16h a = load_a_g(inp, VV, mt * 16, k0, VV);
    v16h b = load_b_h16(wc, LDWC, n0, k0);
    acc = wmma32(a, b, acc);
  }
  const int lane = threadIdx.x & 31;
  const int n = n0 + (lane & 15);
  const int hi = lane >> 4;
  const float bias = bih0[n];
#pragma unroll
  for (int r = 0; r < 8; ++r) {
    const int row = mt * 16 + r + 8 * hi;   // row = b*T + t
    const int b = row >> 10, t = row & 1023;
    gx0[(size_t)(t * BB + b) * G3 + n] = acc[r] + bias;  // time-major for scan
  }
}

// ---------------------------------------------------------------------------
// K3: encoder recurrence. 1 workgroup, 8 waves; per-wave B tiles preloaded.
// LDS (dynamic, 64 KB): h1[64x32] h2[64x32] gA[64x96] gB[64x96]
// ---------------------------------------------------------------------------
__global__ __launch_bounds__(256) void k_enc(
    const float* __restrict__ gx0, const float* __restrict__ whh0,
    const float* __restrict__ wih1, const float* __restrict__ whh1,
    const float* __restrict__ bhh0, const float* __restrict__ bih1,
    const float* __restrict__ bhh1, float* __restrict__ hfin) {
  extern __shared__ char smem[];
  float* h1 = (float*)smem;
  float* h2 = h1 + 2048;
  float* gA = h2 + 2048;   // gx1
  float* gB = gA + 6144;   // gh1 / gh2

  const int tid = threadIdx.x;
  const int wave = tid >> 5;

  int mt[3], n0[3];
  v16h vb_hh0[3], vb_ih1[3], vb_hh1[3];
#pragma unroll
  for (int j = 0; j < 3; ++j) {
    const int tl = wave * 3 + j;            // 24 tiles / 8 waves
    mt[j] = tl / 6; n0[j] = (tl % 6) * 16;
    vb_hh0[j] = load_b_wT(whh0, HH, n0[j], 0);
    vb_ih1[j] = load_b_wT(wih1, HH, n0[j], 0);
    vb_hh1[j] = load_b_wT(whh1, HH, n0[j], 0);
  }

  for (int i = tid; i < 4096; i += 256) h1[i] = 0.0f;  // zeroes h1 and h2
  __syncthreads();

  for (int t = 0; t < TT; ++t) {
    if (t + 1 < TT)  // stream next step's gates into cache (global_prefetch_b8)
      __builtin_prefetch(gx0 + (size_t)(t + 1) * BB * G3 + tid * 24, 0, 0);

    // gh1 = h1 @ Whh0^T
#pragma unroll
    for (int j = 0; j < 3; ++j) {
      v8f acc = {};
      acc = wmma32(load_a_lds(h1, mt[j] * 16), vb_hh0[j], acc);
      store_d_lds(gB, G3, mt[j] * 16, n0[j], acc);
    }
    __syncthreads();

    // combine layer 1 (gx0 already holds bih0)
    const float* gxt = gx0 + (size_t)t * BB * G3;
    for (int e = tid; e < 2048; e += 256) {
      const int b = e >> 5, j = e & 31;
      const float* gxr = gxt + b * G3;
      const float* ghr = gB + b * G3;
      const float r = sigmf(gxr[j] + ghr[j] + bhh0[j]);
      const float z = sigmf(gxr[32 + j] + ghr[32 + j] + bhh0[32 + j]);
      const float nn = tanhf(gxr[64 + j] + r * (ghr[64 + j] + bhh0[64 + j]));
      h1[e] = (1.0f - z) * nn + z * h1[e];
    }
    __syncthreads();

    // gx1 = h1 @ Wih1^T ; gh2 = h2 @ Whh1^T  (independent -> single phase)
#pragma unroll
    for (int j = 0; j < 3; ++j) {
      v8f ax = {};
      ax = wmma32(load_a_lds(h1, mt[j] * 16), vb_ih1[j], ax);
      store_d_lds(gA, G3, mt[j] * 16, n0[j], ax);
      v8f ah = {};
      ah = wmma32(load_a_lds(h2, mt[j] * 16), vb_hh1[j], ah);
      store_d_lds(gB, G3, mt[j] * 16, n0[j], ah);
    }
    __syncthreads();

    // combine layer 2
    for (int e = tid; e < 2048; e += 256) {
      const int b = e >> 5, j = e & 31;
      const float* gxr = gA + b * G3;
      const float* ghr = gB + b * G3;
      const float r = sigmf(gxr[j] + bih1[j] + ghr[j] + bhh1[j]);
      const float z = sigmf(gxr[32 + j] + bih1[32 + j] + ghr[32 + j] + bhh1[32 + j]);
      const float nn =
          tanhf(gxr[64 + j] + bih1[64 + j] + r * (ghr[64 + j] + bhh1[64 + j]));
      h2[e] = (1.0f - z) * nn + z * h2[e];
    }
    __syncthreads();
  }

  for (int i = tid; i < 2048; i += 256) {
    hfin[i] = h1[i];
    hfin[2048 + i] = h2[i];
  }
}

// ---------------------------------------------------------------------------
// K4: decoder scan with greedy argmax feedback. 1 workgroup, 8 waves.
// ---------------------------------------------------------------------------
__global__ __launch_bounds__(256) void k_dec(
    const float* __restrict__ hfin, const float* __restrict__ demb,
    const float* __restrict__ wih0, const float* __restrict__ whh0,
    const float* __restrict__ bih0, const float* __restrict__ bhh0,
    const float* __restrict__ wih1, const float* __restrict__ whh1,
    const float* __restrict__ bih1, const float* __restrict__ bhh1,
    const float* __restrict__ pW1, const float* __restrict__ pb1,
    const float* __restrict__ pW2, const float* __restrict__ pb2,
    float* __restrict__ out, float* __restrict__ wlog) {
  extern __shared__ char smem[];
  float* d1 = (float*)smem;         // 2048
  float* d2 = d1 + 2048;            // 2048
  float* gx = d2 + 2048;            // 6144
  float* gh = gx + 6144;            // 6144
  float* zb = gh + 6144;            // 2048 (pred hidden, ld 32)
  float* rmaxs = zb + 2048;         // 64
  float* rsums = rmaxs + 64;        // 64
  float* smax = rsums + 64;         // 256
  float* ssum = smax + 256;         // 256
  int* simx = (int*)(ssum + 256);   // 256
  int* idxb = simx + 256;           // 64

  const int tid = threadIdx.x;
  const int wave = tid >> 5;

  int mt[3], n0[3];
  v16h wb_ih0[3], wb_hh0[3], wb_ih1[3], wb_hh1[3];
#pragma unroll
  for (int j = 0; j < 3; ++j) {
    const int tl = wave * 3 + j;
    mt[j] = tl / 6; n0[j] = (tl % 6) * 16;
    wb_ih0[j] = load_b_wT(wih0, HH, n0[j], 0);
    wb_hh0[j] = load_b_wT(whh0, HH, n0[j], 0);
    wb_ih1[j] = load_b_wT(wih1, HH, n0[j], 0);
    wb_hh1[j] = load_b_wT(whh1, HH, n0[j], 0);
  }
  const int mtp = wave >> 1;            // pred_W1: 4x2 tiles, one per wave
  const int n0p = (wave & 1) * 16;
  v16h wb_p1 = load_b_wT(pW1, HH, n0p, 0);

  // init: load encoder state, seed idx, emit t=0 (one-hot START)
  for (int i = tid; i < 2048; i += 256) {
    d1[i] = hfin[i];
    d2[i] = hfin[2048 + i];
  }
  if (tid < BB) idxb[tid] = STARTTOK;
  for (int g = tid; g < BB * VV; g += 256) {
    const int b = g / VV, v = g % VV;
    out[(size_t)b * TT * VV + v] = (v == STARTTOK) ? 1.0f : 0.0f;
  }
  if (tid < BB) out[NOUT + (size_t)tid * TT] = (float)STARTTOK;
  __syncthreads();

  const int tg = tid & 3, row = tid >> 2;  // 4 threads per batch row

  for (int step = 1; step < TT; ++step) {
    // layer 1: gx = emb[idx] @ Wih0^T ; gh = d1 @ Whh0^T
#pragma unroll
    for (int j = 0; j < 3; ++j) {
      v8f ax = {};
      ax = wmma32(load_a_gather(demb, idxb, mt[j] * 16), wb_ih0[j], ax);
      store_d_lds(gx, G3, mt[j] * 16, n0[j], ax);
      v8f ah = {};
      ah = wmma32(load_a_lds(d1, mt[j] * 16), wb_hh0[j], ah);
      store_d_lds(gh, G3, mt[j] * 16, n0[j], ah);
    }
    __syncthreads();

    for (int e = tid; e < 2048; e += 256) {
      const int b = e >> 5, j = e & 31;
      const float* gxr = gx + b * G3;
      const float* ghr = gh + b * G3;
      const float r = sigmf(gxr[j] + bih0[j] + ghr[j] + bhh0[j]);
      const float z = sigmf(gxr[32 + j] + bih0[32 + j] + ghr[32 + j] + bhh0[32 + j]);
      const float nn =
          tanhf(gxr[64 + j] + bih0[64 + j] + r * (ghr[64 + j] + bhh0[64 + j]));
      d1[e] = (1.0f - z) * nn + z * d1[e];
    }
    __syncthreads();

    // layer 2: gx = d1 @ Wih1^T ; gh = d2 @ Whh1^T
#pragma unroll
    for (int j = 0; j < 3; ++j) {
      v8f ax = {};
      ax = wmma32(load_a_lds(d1, mt[j] * 16), wb_ih1[j], ax);
      store_d_lds(gx, G3, mt[j] * 16, n0[j], ax);
      v8f ah = {};
      ah = wmma32(load_a_lds(d2, mt[j] * 16), wb_hh1[j], ah);
      store_d_lds(gh, G3, mt[j] * 16, n0[j], ah);
    }
    __syncthreads();

    for (int e = tid; e < 2048; e += 256) {
      const int b = e >> 5, j = e & 31;
      const float* gxr = gx + b * G3;
      const float* ghr = gh + b * G3;
      const float r = sigmf(gxr[j] + bih1[j] + ghr[j] + bhh1[j]);
      const float z = sigmf(gxr[32 + j] + bih1[32 + j] + ghr[32 + j] + bhh1[32 + j]);
      const float nn =
          tanhf(gxr[64 + j] + bih1[64 + j] + r * (ghr[64 + j] + bhh1[64 + j]));
      d2[e] = (1.0f - z) * nn + z * d2[e];
    }
    __syncthreads();

    // pred layer 1: zb = leaky_relu(d2 @ pW1^T + pb1), one tile per wave
    {
      v8f acc = {};
      acc = wmma32(load_a_lds(d2, mtp * 16), wb_p1, acc);
      const int lane = tid & 31;
      const int n = n0p + (lane & 15);
      const int hi = lane >> 4;
      const float bias = pb1[n];
#pragma unroll
      for (int r8 = 0; r8 < 8; ++r8) {
        float v = acc[r8] + bias;
        v = (v > 0.0f) ? v : 0.1f * v;
        zb[(mtp * 16 + r8 + 8 * hi) * HH + n] = v;
      }
    }
    __syncthreads();

    // logits = zb @ pW2^T + pb2 -> ws scratch (ld 400); 100 tiles, EXEC-uniform
    for (int tl = wave; tl < 100; tl += 8) {
      const int mtl = tl / 25, n0l = (tl % 25) * 16;
      v16h a = load_a_lds(zb, mtl * 16);
      v16h b = load_b_wT_ng(pW2, HH, n0l, 0, VV);
      v8f acc = {};
      acc = wmma32(a, b, acc);
      const int lane = tid & 31;
      const int n = n0l + (lane & 15);
      const int hi = lane >> 4;
      if (n < VV) {
        const float bias = pb2[n];
#pragma unroll
        for (int r8 = 0; r8 < 8; ++r8)
          wlog[(mtl * 16 + r8 + 8 * hi) * 400 + n] = acc[r8] + bias;
      }
    }
    __syncthreads();

    // argmax partials (first-max tie rule preserved by increasing-n scan)
    {
      float mv = -1e30f;
      int mi = VV;
      for (int n = tg; n < VV; n += 4) {
        const float l = wlog[row * 400 + n];
        if (l > mv) { mv = l; mi = n; }
      }
      smax[row * 4 + tg] = mv;
      simx[row * 4 + tg] = mi;
    }
    __syncthreads();

    if (tid < BB) {
      float Mv = -1e30f;
      int Mi = VV;
#pragma unroll
      for (int q = 0; q < 4; ++q) {
        const float v = smax[tid * 4 + q];
        const int i = simx[tid * 4 + q];
        if (v > Mv || (v == Mv && i < Mi)) { Mv = v; Mi = i; }
      }
      rmaxs[tid] = Mv;
      idxb[tid] = Mi;                         // greedy feedback
      out[NOUT + (size_t)tid * TT + step] = (float)Mi;
    }
    __syncthreads();

    // softmax denominator
    {
      float s = 0.0f;
      const float Mv = rmaxs[row];
      for (int n = tg; n < VV; n += 4) s += __expf(wlog[row * 400 + n] - Mv);
      ssum[row * 4 + tg] = s;
    }
    __syncthreads();

    if (tid < BB)
      rsums[tid] = 1.0f / (ssum[tid * 4] + ssum[tid * 4 + 1] +
                           ssum[tid * 4 + 2] + ssum[tid * 4 + 3]);
    __syncthreads();

    // write probs for this step
    {
      const float Mv = rmaxs[row];
      const float is = rsums[row];
      float* orow = out + (size_t)row * TT * VV + (size_t)step * VV;
      for (int n = tg; n < VV; n += 4)
        orow[n] = __expf(wlog[row * 400 + n] - Mv) * is;
    }
    __syncthreads();
  }
}

// ---------------------------------------------------------------------------
extern "C" void kernel_launch(void* const* d_in, const int* in_sizes, int n_in,
                              void* d_out, int out_size, void* d_ws,
                              size_t ws_size, hipStream_t stream) {
  (void)in_sizes; (void)n_in; (void)out_size; (void)ws_size;

  const float* input     = (const float*)d_in[0];
  const float* enc_emb_W = (const float*)d_in[1];
  const float* enc_Wih0  = (const float*)d_in[2];
  const float* enc_Whh0  = (const float*)d_in[3];
  const float* enc_bih0  = (const float*)d_in[4];
  const float* enc_bhh0  = (const float*)d_in[5];
  const float* enc_Wih1  = (const float*)d_in[6];
  const float* enc_Whh1  = (const float*)d_in[7];
  const float* enc_bih1  = (const float*)d_in[8];
  const float* enc_bhh1  = (const float*)d_in[9];
  const float* dec_emb   = (const float*)d_in[10];
  const float* dec_Wih0  = (const float*)d_in[11];
  const float* dec_Whh0  = (const float*)d_in[12];
  const float* dec_bih0  = (const float*)d_in[13];
  const float* dec_bhh0  = (const float*)d_in[14];
  const float* dec_Wih1  = (const float*)d_in[15];
  const float* dec_Whh1  = (const float*)d_in[16];
  const float* dec_bih1  = (const float*)d_in[17];
  const float* dec_bhh1  = (const float*)d_in[18];
  const float* pred_W1   = (const float*)d_in[19];
  const float* pred_b1   = (const float*)d_in[20];
  const float* pred_W2   = (const float*)d_in[21];
  const float* pred_b2   = (const float*)d_in[22];
  float* out = (float*)d_out;

  // workspace layout (offsets 256B-aligned)
  char* w = (char*)d_ws;
  _Float16* wc = (_Float16*)w;                                   // 96*416*2
  size_t off = (size_t)G3 * LDWC * 2;                            // 79,872
  float* gx0 = (float*)(w + off);                                // 24 MB
  off += (size_t)TT * BB * G3 * 4;
  float* hfin = (float*)(w + off);                               // 16 KB
  off += 4096 * 4;
  float* wlog = (float*)(w + off);                               // 100 KB

  // K1: fold embedding into layer-0 input weight (f16)
  k_wc<<<(G3 * LDWC + 255) / 256, 256, 0, stream>>>(enc_Wih0, enc_emb_W, wc);

  // K2: all-timestep layer-0 input gates, WMMA GEMM (24576 waves)
  k_gx0<<<6144, 128, 0, stream>>>(input, wc, enc_bih0, gx0);

  // K3: encoder scan (1 persistent workgroup, 64 KB dynamic LDS)
  k_enc<<<1, 256, 16384 * 4, stream>>>(gx0, enc_Whh0, enc_Wih1, enc_Whh1,
                                       enc_bhh0, enc_bih1, enc_bhh1, hfin);

  // K4: decoder scan (1 persistent workgroup, ~76 KB dynamic LDS)
  const size_t dec_smem = (2048 * 3 + 6144 * 2 + 64 + 64 + 256 + 256) * 4 +
                          (256 + 64) * 4;
  k_dec<<<1, 256, dec_smem, stream>>>(hfin, dec_emb, dec_Wih0, dec_Whh0,
                                      dec_bih0, dec_bhh0, dec_Wih1, dec_Whh1,
                                      dec_bih1, dec_bhh1, pred_W1, pred_b1,
                                      pred_W2, pred_b2, out, wlog);
}
